// PoS_GAT_70308614635893
// MI455X (gfx1250) — matrix operand order repaired and checked by reading, not compiled, not run on hardware
//
#include <hip/hip_runtime.h>
#include <hip/hip_bf16.h>

// Problem constants (from reference)
#define B_   128
#define L_   256
#define D_   512
#define NL_  6
#define NEGV (-9.0e15f)

typedef __attribute__((ext_vector_type(16))) __bf16 v16bf;
typedef __attribute__((ext_vector_type(8)))  float  v8f;
typedef int v4i_g __attribute__((vector_size(16)));   // matches builtin pointee type

static __device__ __forceinline__ __bf16 bfc(float f) { return (__bf16)f; }

// ---------------------------------------------------------------------------
// CDNA5 async LDS copy path (GLOBAL_LOAD_ASYNC_TO_LDS_B128 / ASYNCcnt),
// guarded so a missing builtin falls back to plain copies and still compiles.
// Probe-learned signature: (v4i AS1*, v4i AS3*, imm i32 offset, imm i32 cpol).
// ---------------------------------------------------------------------------
#if defined(__has_builtin)
#  if __has_builtin(__builtin_amdgcn_global_load_async_to_lds_b128)
#    define ASYNC_LDS 1
#  endif
#endif
#ifndef ASYNC_LDS
#  define ASYNC_LDS 0
#endif

#if ASYNC_LDS
__device__ __forceinline__ void async_b128(const float* g, float* l) {
    __builtin_amdgcn_global_load_async_to_lds_b128(
        (__attribute__((address_space(1))) v4i_g*)(unsigned long long)g,
        (__attribute__((address_space(3))) v4i_g*)(unsigned int)(unsigned long long)l,
        0, 0);
}
#  if __has_builtin(__builtin_amdgcn_s_wait_asynccnt)
#    define WAIT_ASYNC(N) __builtin_amdgcn_s_wait_asynccnt(N)
#  else
#    define WAIT_ASYNC_STR(N) "s_wait_asynccnt " #N
#    define WAIT_ASYNC(N) asm volatile(WAIT_ASYNC_STR(N) ::: "memory")
#  endif
#  define WAIT_DS0() asm volatile("s_wait_dscnt 0x0" ::: "memory")
#else
__device__ __forceinline__ void async_b128(const float* g, float* l) {
    *(float4*)l = *(const float4*)g;
}
#  define WAIT_ASYNC(N) do {} while (0)
#  define WAIT_DS0()    do {} while (0)
#endif

__device__ __forceinline__ float pos_w(int j, int l, int al, int tl) {
    float ctx = (float)(tl - al);
    int r = l + al - 1;
    if (j < l)  return 1.0f - (float)(l - j) / ctx;
    if (j <= r) return 0.0f;
    if (j < tl) return 1.0f - ((float)j - (float)r) / ctx;
    return 0.0f;
}

// ---------------------------------------------------------------------------
// K0: per-layer scalar coefs s0 = sum(a[0:256]), s1 = sum(a[256:512])
// ---------------------------------------------------------------------------
__global__ __launch_bounds__(256) void k_coef(const float* __restrict__ a_stack,
                                              float* __restrict__ coef) {
    int layer = blockIdx.x;
    const float* a = a_stack + (size_t)layer * 2 * D_;
    __shared__ float red[256];
    int t = threadIdx.x;
    float v0 = a[t];
    float v1 = a[256 + t];
    red[t] = v0; __syncthreads();
    for (int s = 128; s > 0; s >>= 1) { if (t < s) red[t] += red[t + s]; __syncthreads(); }
    if (t == 0) coef[layer * 2 + 0] = red[0];
    __syncthreads();
    red[t] = v1; __syncthreads();
    for (int s = 128; s > 0; s >>= 1) { if (t < s) red[t] += red[t + s]; __syncthreads(); }
    if (t == 0) coef[layer * 2 + 1] = red[0];
}

// ---------------------------------------------------------------------------
// K1: h[b] = A[b] @ W  (A = pos-weighted x for layer 0, else h1)
// bf16 WMMA 16x16x32, f32 accum. 256 thr (8 waves), 64x64 C tile,
// double-buffered async LDS staging of raw f32 panels.
// ---------------------------------------------------------------------------
__global__ __launch_bounds__(256) void k_gemm_xw(
    const float* __restrict__ src, const float* __restrict__ Wl,
    float* __restrict__ out,
    const int* __restrict__ left, const int* __restrict__ alen,
    const int* __restrict__ tlen, int apply_posw)
{
    const int b    = blockIdx.z;
    const int Mt   = blockIdx.y * 64;
    const int Nt   = blockIdx.x * 64;
    const int t    = threadIdx.x;
    const int w    = t >> 5;
    const int lane = t & 31;
    const int wm   = w >> 2;
    const int wn   = w & 3;

    __shared__ float shA[2][64][32];   // raw f32 A panel (x or h1)
    __shared__ float shB[2][32][64];   // raw f32 W panel
    __shared__ float rowW[64];

    const float* srcb = src + (size_t)b * L_ * D_;
    if (t < 64) rowW[t] = apply_posw ? pos_w(Mt + t, left[b], alen[b], tlen[b]) : 1.0f;
    __syncthreads();

    const int ar = t >> 2, ac = (t & 3) * 8;   // A: 2 x b128 per thread
    const int br = t >> 3, bc = (t & 7) * 8;   // B: 2 x b128 per thread

#define ISSUE(bu, kk) do {                                              \
        const float* pa = srcb + (size_t)(Mt + ar) * D_ + (kk) + ac;    \
        async_b128(pa,     &shA[bu][ar][ac]);                           \
        async_b128(pa + 4, &shA[bu][ar][ac + 4]);                       \
        const float* pb = Wl + (size_t)((kk) + br) * D_ + Nt + bc;      \
        async_b128(pb,     &shB[bu][br][bc]);                           \
        async_b128(pb + 4, &shB[bu][br][bc + 4]);                       \
    } while (0)

    ISSUE(0, 0);

    v8f acc0 = {}, acc1 = {};
    const int mrow = lane & 15;
    const int khA  = (lane < 16) ? 0 : 8;   // A K-halves per ISA layout
    const int khB  = (lane < 16) ? 0 : 8;   // B K-halves per ISA layout
    const int ncol = wn * 16 + (lane & 15);
    const float pw0 = rowW[wm * 32 + mrow];
    const float pw1 = rowW[wm * 32 + 16 + mrow];

    int cur = 0;
    for (int kk = 0; kk < D_; kk += 32) {
        if (kk + 32 < D_) {
            ISSUE(cur ^ 1, kk + 32);    // prefetch next chunk into other buffer
            WAIT_ASYNC(0x4);            // previous 4 async ops (this chunk) landed
        } else {
            WAIT_ASYNC(0x0);
        }
#if !ASYNC_LDS
        __builtin_prefetch(Wl + (size_t)((kk + 32) + br) * D_ + Nt + bc, 0, 1);
#endif
        __syncthreads();

        v16bf bfrag, af0, af1;
        #pragma unroll
        for (int v = 0; v < 8; ++v) {
            bfrag[2 * v]     = bfc(shB[cur][khB + 2 * v + (v >= 4 ? 8 : 0)][ncol]);
            bfrag[2 * v + 1] = bfc(shB[cur][khB + 2 * v + 1 + (v >= 4 ? 8 : 0)][ncol]);
        }
        #pragma unroll
        for (int v = 0; v < 4; ++v) {
            af0[2 * v]         = bfc(shA[cur][wm * 32 + mrow][khA + 2 * v] * pw0);
            af0[2 * v + 1]     = bfc(shA[cur][wm * 32 + mrow][khA + 2 * v + 1] * pw0);
            af0[8 + 2 * v]     = bfc(shA[cur][wm * 32 + mrow][16 + khA + 2 * v] * pw0);
            af0[8 + 2 * v + 1] = bfc(shA[cur][wm * 32 + mrow][16 + khA + 2 * v + 1] * pw0);
            af1[2 * v]         = bfc(shA[cur][wm * 32 + 16 + mrow][khA + 2 * v] * pw1);
            af1[2 * v + 1]     = bfc(shA[cur][wm * 32 + 16 + mrow][khA + 2 * v + 1] * pw1);
            af1[8 + 2 * v]     = bfc(shA[cur][wm * 32 + 16 + mrow][16 + khA + 2 * v] * pw1);
            af1[8 + 2 * v + 1] = bfc(shA[cur][wm * 32 + 16 + mrow][16 + khA + 2 * v + 1] * pw1);
        }
        acc0 = __builtin_amdgcn_wmma_f32_16x16x32_bf16(false, af0, false, bfrag,
                                                       (short)0, acc0, false, false);
        acc1 = __builtin_amdgcn_wmma_f32_16x16x32_bf16(false, af1, false, bfrag,
                                                       (short)0, acc1, false, false);
        __syncthreads();
        cur ^= 1;
    }
#undef ISSUE

    float* outb = out + (size_t)b * L_ * D_;
    const int rbase = (lane < 16) ? 0 : 8;
    const int col   = Nt + wn * 16 + (lane & 15);
    #pragma unroll
    for (int r = 0; r < 8; ++r) {
        outb[(size_t)(Mt + wm * 32 + rbase + r) * D_ + col]      = acc0[r];
        outb[(size_t)(Mt + wm * 32 + 16 + rbase + r) * D_ + col] = acc1[r];
    }
}

// ---------------------------------------------------------------------------
// K2: g[b,j] = h[b,j,:] . a2
// ---------------------------------------------------------------------------
__global__ __launch_bounds__(256) void k_gvec(const float* __restrict__ h,
                                              const float* __restrict__ a_layer,
                                              float* __restrict__ g)
{
    const float* a2 = a_layer + D_;
    __shared__ float sa[D_];
    int t = threadIdx.x;
    sa[t] = a2[t]; sa[t + 256] = a2[t + 256];
    __syncthreads();
    size_t row = (size_t)blockIdx.x * 256 + t;
    const float* hr = h + row * D_;
    float s = 0.f;
    #pragma unroll 4
    for (int k = 0; k < D_; k += 4) {
        float4 hv = *(const float4*)(hr + k);
        s += hv.x * sa[k] + hv.y * sa[k + 1] + hv.z * sa[k + 2] + hv.w * sa[k + 3];
    }
    g[row] = s;
}

// ---------------------------------------------------------------------------
// K3: attention strip (16 rows) — logits -> masked softmax -> att@h via WMMA.
// LDS overlay: the f32 h-row strip (logit phase) aliases the f32 B-panel
// chunk (GEMM phase). B-panel staging is wave-private (wave w only reads its
// own 64 columns) -> per-wave async copies, no block barriers in the K-loop.
// mode 0: dst = posw * relu(hp); mode 1: dst = relu(hp); mode 2: dst += relu.
// ---------------------------------------------------------------------------
__global__ __launch_bounds__(256) void k_att(
    const float* __restrict__ h, const float* __restrict__ g,
    const int* __restrict__ adj, const float* __restrict__ coef, int layer,
    const int* __restrict__ left, const int* __restrict__ alen,
    const int* __restrict__ tlen,
    float* __restrict__ dst, int mode)
{
    const int b      = blockIdx.y;
    const int istrip = blockIdx.x;
    const int i0     = istrip * 16;
    const int t      = threadIdx.x;
    const int w      = t >> 5;
    const int lane   = t & 31;

    // LDS overlay: [att 16x256 | g 256 | red 16x16 | rowstat 16 | ovl 32x512]
    __shared__ __align__(16) float smem[4624 + 32 * D_];
    float* sh_att  = smem;          // 16*256
    float* sh_g    = smem + 4096;   // 256
    float* redm    = smem + 4352;   // 16*16
    float* rowstat = smem + 4608;   // 16
    float* sh_h    = smem + 4624;   // 16*512 f32 (phase 1-2)
    float* chunk   = smem + 4624;   // 32*512 f32 (phase 3, aliases sh_h)

    const float s0 = coef[layer * 2], s1 = coef[layer * 2 + 1];
    const float* hb = h + (size_t)b * L_ * D_;

    { // async-stage this strip's 16 h rows (full f32 for logits)
        const float* p = hb + (size_t)(i0 + (t >> 4)) * D_ + (t & 15) * 32;
        float* q = sh_h + (size_t)(t >> 4) * D_ + (t & 15) * 32;
        #pragma unroll
        for (int u = 0; u < 8; ++u) async_b128(p + u * 4, q + u * 4);
    }
    sh_g[t] = g[(size_t)b * L_ + t];
    WAIT_ASYNC(0x0);
    __syncthreads();

    // logits + masked softmax: 16 threads cooperate per row
    const int row = t & 15, sub = t >> 4;
    const int j0 = sub * 16;
    const int* adjr = adj + ((size_t)b * L_ + (i0 + row)) * L_;
    float mx = NEGV;
    #pragma unroll
    for (int u = 0; u < 16; ++u) {
        int j = j0 + u;
        float v = s0 * sh_h[row * D_ + 2 * j] + s1 * sh_h[row * D_ + 2 * j + 1] + sh_g[j];
        v = fmaxf(v, 0.f);
        v = (adjr[j] > 0) ? v : NEGV;
        sh_att[row * L_ + j] = v;
        mx = fmaxf(mx, v);
    }
    redm[row * 16 + sub] = mx;
    __syncthreads();
    if (sub == 0) {
        float m = redm[row * 16];
        #pragma unroll
        for (int u = 1; u < 16; ++u) m = fmaxf(m, redm[row * 16 + u]);
        rowstat[row] = m;
    }
    __syncthreads();
    float rmax = rowstat[row];
    float ls = 0.f;
    #pragma unroll
    for (int u = 0; u < 16; ++u) {
        int j = j0 + u;
        float v = sh_att[row * L_ + j];
        float e = (v <= 0.5f * NEGV) ? 0.f : __expf(v - rmax);
        sh_att[row * L_ + j] = e;
        ls += e;
    }
    redm[row * 16 + sub] = ls;
    __syncthreads();
    if (sub == 0) {
        float s = 0.f;
        #pragma unroll
        for (int u = 0; u < 16; ++u) s += redm[row * 16 + u];
        rowstat[row] = 1.0f / s;
    }
    __syncthreads();
    float rinv = rowstat[row];
    #pragma unroll
    for (int u = 0; u < 16; ++u) sh_att[row * L_ + j0 + u] *= rinv;
    __syncthreads();   // sh_h is dead past this point; chunk may overwrite it

    // hp(16x512) = att(16x256) @ h[b](256x512); wave w owns cols [64w, 64w+64)
    v8f acc[4] = {};
    const int mrow = lane & 15;
    const int khA  = (lane < 16) ? 0 : 8;
    const int khB  = (lane < 16) ? 0 : 8;
    for (int ck = 0; ck < L_; ck += 32) {
        { // wave-private staging: lane stages row ck+lane, cols [64w, 64w+64)
            const float* p = hb + (size_t)(ck + lane) * D_ + w * 64;
            float* q = chunk + (size_t)lane * D_ + w * 64;
            #pragma unroll
            for (int u = 0; u < 16; ++u) async_b128(p + u * 4, q + u * 4);
        }
        WAIT_ASYNC(0x0);   // wave-level only; no block barrier needed

        v16bf afrag;
        #pragma unroll
        for (int v = 0; v < 4; ++v) {
            afrag[2 * v]         = bfc(sh_att[mrow * L_ + ck + khA + 2 * v]);
            afrag[2 * v + 1]     = bfc(sh_att[mrow * L_ + ck + khA + 2 * v + 1]);
            afrag[8 + 2 * v]     = bfc(sh_att[mrow * L_ + ck + 16 + khA + 2 * v]);
            afrag[8 + 2 * v + 1] = bfc(sh_att[mrow * L_ + ck + 16 + khA + 2 * v + 1]);
        }
        #pragma unroll
        for (int nt = 0; nt < 4; ++nt) {
            int col = w * 64 + nt * 16 + (lane & 15);
            v16bf bfrag;
            #pragma unroll
            for (int v = 0; v < 8; ++v) {
                bfrag[2 * v]     = bfc(chunk[(khB + 2 * v + (v >= 4 ? 8 : 0)) * D_ + col]);
                bfrag[2 * v + 1] = bfc(chunk[(khB + 2 * v + 1 + (v >= 4 ? 8 : 0)) * D_ + col]);
            }
            acc[nt] = __builtin_amdgcn_wmma_f32_16x16x32_bf16(false, afrag, false, bfrag,
                                                              (short)0, acc[nt], false, false);
        }
        WAIT_DS0();   // our LDS reads done before next iteration's async overwrite
    }

    // epilogue
    float* db = dst + (size_t)b * L_ * D_;
    const int rbase = (lane < 16) ? 0 : 8;
    const int lb = left[b], ab = alen[b], tb = tlen[b];
    #pragma unroll
    for (int nt = 0; nt < 4; ++nt) {
        int col = w * 64 + nt * 16 + (lane & 15);
        #pragma unroll
        for (int r = 0; r < 8; ++r) {
            int gr = i0 + rbase + r;
            float v = fmaxf(acc[nt][r], 0.f);   // relu(elu(x)) == relu(x)
            size_t idx = (size_t)gr * D_ + col;
            if (mode == 0)      db[idx] = pos_w(gr, lb, ab, tb) * v;
            else if (mode == 1) db[idx] = v;
            else                db[idx] += v;
        }
    }
}

// ---------------------------------------------------------------------------
// K4a: s[b,d] = 0.2 * sum over aspect rows of x_graph
// ---------------------------------------------------------------------------
__global__ __launch_bounds__(256) void k_svec(const float* __restrict__ xg,
                                              const int* __restrict__ left,
                                              const int* __restrict__ alen,
                                              float* __restrict__ s)
{
    int b = blockIdx.x, t = threadIdx.x;
    int l = left[b], a = alen[b];
    const float* xb = xg + (size_t)b * L_ * D_;
    #pragma unroll
    for (int u = 0; u < 2; ++u) {
        int d = t + u * 256;
        float acc = 0.f;
        for (int r = 0; r < a; ++r) acc += xb[(size_t)(l + r) * D_ + d];
        s[(size_t)b * D_ + d] = 0.2f * acc;
    }
}

// ---------------------------------------------------------------------------
// K4b: t[m] = s[b].x[b,m]; alpha = softmax(t); out[b] = sum_m alpha[m] x[b,m]
// ---------------------------------------------------------------------------
__global__ __launch_bounds__(256) void k_final(const float* __restrict__ x,
                                               const float* __restrict__ s,
                                               float* __restrict__ out)
{
    int b = blockIdx.x, t = threadIdx.x;
    __shared__ float ss[D_];
    __shared__ float alpha[L_];
    __shared__ float red[256];
    ss[t] = s[(size_t)b * D_ + t];
    ss[t + 256] = s[(size_t)b * D_ + t + 256];
    __syncthreads();
    const float* xb = x + (size_t)b * L_ * D_;
    const float* xr = xb + (size_t)t * D_;
    float dot = 0.f;
    #pragma unroll 4
    for (int k = 0; k < D_; k += 4) {
        float4 v = *(const float4*)(xr + k);
        dot += v.x * ss[k] + v.y * ss[k + 1] + v.z * ss[k + 2] + v.w * ss[k + 3];
    }
    red[t] = dot; __syncthreads();
    for (int st = 128; st > 0; st >>= 1) { if (t < st) red[t] = fmaxf(red[t], red[t + st]); __syncthreads(); }
    float mx = red[0];
    __syncthreads();
    float e = __expf(dot - mx);
    red[t] = e; __syncthreads();
    for (int st = 128; st > 0; st >>= 1) { if (t < st) red[t] += red[t + st]; __syncthreads(); }
    float inv = 1.f / red[0];
    alpha[t] = e * inv;
    __syncthreads();
    #pragma unroll
    for (int u = 0; u < 2; ++u) {
        int d = t + u * 256;
        float acc = 0.f;
        for (int m = 0; m < L_; ++m) acc += alpha[m] * xb[(size_t)m * D_ + d];
        out[(size_t)b * D_ + d] = acc;
    }
}

// ---------------------------------------------------------------------------
extern "C" void kernel_launch(void* const* d_in, const int* in_sizes, int n_in,
                              void* d_out, int out_size, void* d_ws, size_t ws_size,
                              hipStream_t stream) {
    const float* x    = (const float*)d_in[0];
    const float* Ws   = (const float*)d_in[1];
    const float* As   = (const float*)d_in[2];
    const int*   adj  = (const int*)d_in[3];
    const int*   left = (const int*)d_in[4];
    const int*   alen = (const int*)d_in[5];
    const int*   tlen = (const int*)d_in[6];
    float* out = (float*)d_out;

    // workspace layout (floats): h | h1 | x_graph | g | s | coef  (~192.5 MB)
    float* ws   = (float*)d_ws;
    const size_t BLD = (size_t)B_ * L_ * D_;
    float* h    = ws;
    float* h1   = h + BLD;
    float* xg   = h1 + BLD;
    float* g    = xg + BLD;
    float* sv   = g + (size_t)B_ * L_;
    float* coef = sv + (size_t)B_ * D_;

    k_coef<<<NL_, 256, 0, stream>>>(As, coef);

    for (int layer = 0; layer < NL_; ++layer) {
        const float* src = (layer == 0) ? x : h1;
        k_gemm_xw<<<dim3(D_ / 64, L_ / 64, B_), 256, 0, stream>>>(
            src, Ws + (size_t)layer * D_ * D_, h, left, alen, tlen, layer == 0 ? 1 : 0);
        k_gvec<<<(B_ * L_) / 256, 256, 0, stream>>>(h, As + (size_t)layer * 2 * D_, g);
        int mode   = (layer == 0) ? 0 : ((layer == 1) ? 1 : 2);
        float* dst = (layer == 0) ? h1 : xg;
        k_att<<<dim3(L_ / 16, B_), 256, 0, stream>>>(h, g, adj, coef, layer,
                                                     left, alen, tlen, dst, mode);
    }
    k_svec<<<B_, 256, 0, stream>>>(xg, left, alen, sv);
    k_final<<<B_, 256, 0, stream>>>(x, sv, out);
}
